// Pairformer_1726576853223
// MI455X (gfx1250) — matrix operand order repaired and checked by reading, not compiled, not run on hardware
//
#include <hip/hip_runtime.h>

// =============================================================================
// Pairformer (2 layers, B=1, N=256, c_s=384, c_z=128) for MI455X / gfx1250.
// All matmuls/einsums/attention run on v_wmma_f32_16x16x32_bf16 (wave32).
// Tiles staged memory->LDS with global_load_async_to_lds_b128 (ASYNCcnt),
// fragments built with ds_load_b128. Hot K-loop is branch-light: all guards
// and addresses are hoisted; K-tails handled in a separate guarded epilogue.
// =============================================================================

typedef __attribute__((ext_vector_type(16))) __bf16 v16bf;
typedef __attribute__((ext_vector_type(8)))  float  v8f;

union BF16Frag { v16bf v; unsigned short u[16]; };
union LdU { uint4 q; unsigned short h[8]; };

__device__ __forceinline__ unsigned short f2bf(float x) {
  union { float f; unsigned u; } c; c.f = x;
  unsigned r = c.u + 0x7FFFu + ((c.u >> 16) & 1u);
  return (unsigned short)(r >> 16);
}
__device__ __forceinline__ float bf2f(unsigned short h) {
  union { unsigned u; float f; } c; c.u = ((unsigned)h) << 16;
  return c.f;
}
__device__ __forceinline__ float sigm(float x) { return 1.0f / (1.0f + __expf(-x)); }

// gfx1250 async global->LDS copy (VGLOBAL opcode 98, tracked by ASYNCcnt).
// addrspace(3) pointers are 32-bit, so the cast yields the LDS byte offset.
__device__ __forceinline__ void async_ld_b128(void* lds, const unsigned short* gsrc) {
  unsigned loff = (unsigned)(unsigned long long)lds;
  asm volatile("global_load_async_to_lds_b128 %0, %1, off"
               :: "v"(loff), "v"(gsrc) : "memory");
}
__device__ __forceinline__ void wait_async0() {
  asm volatile("s_wait_asynccnt 0" ::: "memory");
}

// ------------------------- generic strided batched WMMA GEMM -----------------
// C[b][m][n] = alpha * sum_k A[b][m][k] * B[b][k][n]  (+ bias[n]) (+= if epi)
// Batch z decomposes as b1 = z / bmod, b2 = z % bmod, independent strides.
// AP: 0 => aK==1 (k-contig A), 1 => aM==1 (m-contig A)
// BP: 0 => bK==1 (k-contig B), 1 => bN==1 (n-contig B)
struct GemmP {
  const unsigned short* A;
  const unsigned short* B;
  float* C;
  const float* bias;
  long long aM, aK, bK, bN, cM, cN;
  long long aB1, aB2, bB1, bB2, cB1, cB2;
  int M, N, K, bmod, epi;
  float alpha;
};

template <int AP, int BP>
__global__ __launch_bounds__(256) void wmma_gemm_kernel(GemmP g) {
  // A tile [64 x 32] as As[m][k]; B tile [32 x 128] stored transposed Bt[n][k].
  // Row stride 40 halves = 80B keeps every 8-half chunk 16B aligned.
  __shared__ __align__(16) unsigned short As[64][40];
  __shared__ __align__(16) unsigned short Bt[128][40];

  const int tid  = threadIdx.x;
  const int lane = tid & 31;
  const int wave = tid >> 5;
  const int bm = blockIdx.y * 64;
  const int bn = blockIdx.x * 128;
  const int b1 = (int)blockIdx.z / g.bmod;
  const int b2 = (int)blockIdx.z % g.bmod;
  const long long aOff = (long long)b1 * g.aB1 + (long long)b2 * g.aB2;
  const long long bOff = (long long)b1 * g.bB1 + (long long)b2 * g.bB2;
  const long long cOff = (long long)b1 * g.cB1 + (long long)b2 * g.cB2;

  // wave tile: rows (wave&3)*16, cols (wave>>2)*64 (four 16-wide WMMA tiles)
  const int waveM = (wave & 3) << 4;
  const int waveN = (wave >> 2) << 6;

  v8f acc[4];
#pragma unroll
  for (int t = 0; t < 4; ++t)
#pragma unroll
    for (int r = 0; r < 8; ++r) acc[t][r] = 0.0f;

  // -------- k-invariant staging state (hoisted out of the K loop) ----------
  // A, AP==0: one b128 per thread along k.
  const int aM0 = tid >> 2, aKb = (tid & 3) << 3;        // AP==0 coords
  const int aKt = tid >> 3, aMb = (tid & 7) << 3;        // AP==1 coords
  bool aOk = false, aVec = false; int aRem = 0;
  const unsigned short* aP = nullptr;
  if (AP == 0) {
    aOk = (bm + aM0) < g.M;
    aP  = g.A + aOff + (long long)(bm + aM0) * g.aM + aKb;
  } else {
    aRem = g.M - (bm + aMb);
    aVec = (aRem >= 8);
    aP   = g.A + aOff + (long long)aKt * g.aK + bm + aMb;
  }
  // B, two chunks of 256 threads each.
  int bN0[2], bKb[2], bKt[2], bNb[2], bRem[2];
  bool bOk[2], bVec[2];
  const unsigned short* bP[2];
#pragma unroll
  for (int c = 0; c < 2; ++c) {
    int idx = tid + (c << 8);
    if (BP == 0) {
      bN0[c] = idx >> 2; bKb[c] = (idx & 3) << 3;
      bOk[c] = (bn + bN0[c]) < g.N;
      bP[c]  = g.B + bOff + (long long)(bn + bN0[c]) * g.bN + bKb[c];
      bKt[c] = 0; bNb[c] = 0; bRem[c] = 0; bVec[c] = false;
    } else {
      bKt[c] = idx >> 4; bNb[c] = (idx & 15) << 3;
      bRem[c] = g.N - (bn + bNb[c]);
      bVec[c] = (bRem[c] >= 8);
      bP[c]  = g.B + bOff + (long long)bKt[c] * g.bK + bn + bNb[c];
      bN0[c] = 0; bKb[c] = 0; bOk[c] = false;
    }
  }

  // One-time zero-fill so async-skipped OOB rows read as zero every iteration.
  if ((AP == 0 && bm + 64 > g.M) || (BP == 0 && bn + 128 > g.N)) {
    uint4 z4 = {};
    for (int i = tid; i < 320; i += 256) ((uint4*)&As[0][0])[i] = z4;
    for (int i = tid; i < 640; i += 256) ((uint4*)&Bt[0][0])[i] = z4;
    __syncthreads();
  }

  // fragment read addresses (k-invariant)
  const int am = waveM + (lane & 15);
  const int kg = (lane >> 4) << 3;
  const int kb = (lane >> 4) << 4;

  auto compute_step = [&]() {
    BF16Frag fa;
    *(uint4*)&fa.u[0] = *(const uint4*)&As[am][kg];
    *(uint4*)&fa.u[8] = *(const uint4*)&As[am][kg + 16];
#pragma unroll
    for (int t = 0; t < 4; ++t) {
      int bnn = waveN + (t << 4) + (lane & 15);
      BF16Frag fb;
      *(uint4*)&fb.u[0] = *(const uint4*)&Bt[bnn][kg];
      *(uint4*)&fb.u[8] = *(const uint4*)&Bt[bnn][kg + 16];
      acc[t] = __builtin_amdgcn_wmma_f32_16x16x32_bf16(false, fa.v, false, fb.v,
                                                       (short)0, acc[t], false, false);
    }
  };

  // ---------------- main loop: full K tiles, branch-light ----------------
  const int Kfull = g.K & ~31;
  for (int k0 = 0; k0 < Kfull; k0 += 32) {
    if (AP == 0) {
      if (aOk) async_ld_b128(&As[aM0][aKb], aP);
      aP += 32;
    } else {
      LdU v = {};
      if (aVec) v.q = *(const uint4*)aP;
      else {
#pragma unroll
        for (int j = 0; j < 8; ++j) if (j < aRem) v.h[j] = aP[j];
      }
#pragma unroll
      for (int j = 0; j < 8; ++j) As[aMb + j][aKt] = v.h[j];
      aP += (long long)g.aK * 32;
    }
#pragma unroll
    for (int c = 0; c < 2; ++c) {
      if (BP == 0) {
        if (bOk[c]) async_ld_b128(&Bt[bN0[c]][bKb[c]], bP[c]);
        bP[c] += 32;
      } else {
        LdU v = {};
        if (bVec[c]) v.q = *(const uint4*)bP[c];
        else {
#pragma unroll
          for (int j = 0; j < 8; ++j) if (j < bRem[c]) v.h[j] = bP[c][j];
        }
#pragma unroll
        for (int j = 0; j < 8; ++j) Bt[bNb[c] + j][bKt[c]] = v.h[j];
        bP[c] += (long long)g.bK * 32;
      }
    }
    if (AP == 0 || BP == 0) wait_async0();
    __syncthreads();
    compute_step();
    __syncthreads();
  }

  // ---------------- K tail (guarded, runs at most once) ----------------
  if (g.K & 31) {
    const int k0 = Kfull;
    if (AP == 0) {
      LdU v = {};
      if (aOk) {
#pragma unroll
        for (int j = 0; j < 8; ++j) if (k0 + aKb + j < g.K) v.h[j] = aP[j];
      }
      *(uint4*)&As[aM0][aKb] = v.q;
    } else {
      LdU v = {};
      if (k0 + aKt < g.K) {
        if (aVec) v.q = *(const uint4*)aP;
        else {
#pragma unroll
          for (int j = 0; j < 8; ++j) if (j < aRem) v.h[j] = aP[j];
        }
      }
#pragma unroll
      for (int j = 0; j < 8; ++j) As[aMb + j][aKt] = v.h[j];
    }
#pragma unroll
    for (int c = 0; c < 2; ++c) {
      if (BP == 0) {
        LdU v = {};
        if (bOk[c]) {
#pragma unroll
          for (int j = 0; j < 8; ++j) if (k0 + bKb[c] + j < g.K) v.h[j] = bP[c][j];
        }
        *(uint4*)&Bt[bN0[c]][bKb[c]] = v.q;
      } else {
        LdU v = {};
        if (k0 + bKt[c] < g.K) {
          if (bVec[c]) v.q = *(const uint4*)bP[c];
          else {
#pragma unroll
            for (int j = 0; j < 8; ++j) if (j < bRem[c]) v.h[j] = bP[c][j];
          }
        }
#pragma unroll
        for (int j = 0; j < 8; ++j) Bt[bNb[c] + j][bKt[c]] = v.h[j];
      }
    }
    __syncthreads();
    compute_step();
    __syncthreads();
  }

  // epilogue: C layout — lane L holds column N=L&15, rows r + 8*(L>>4)
  const int mbase = bm + waveM + ((lane >> 4) << 3);
#pragma unroll
  for (int t = 0; t < 4; ++t) {
    int n = bn + waveN + (t << 4) + (lane & 15);
    if (n >= g.N) continue;
    float bv = g.bias ? g.bias[n] : 0.0f;
#pragma unroll
    for (int r = 0; r < 8; ++r) {
      int m = mbase + r;
      if (m >= g.M) continue;
      long long idx = cOff + (long long)m * g.cM + (long long)n * g.cN;
      float val = acc[t][r] * g.alpha + bv;
      if (g.epi) g.C[idx] += val; else g.C[idx] = val;
    }
  }
}

// ------------------------------ elementwise ----------------------------------
__global__ void f32_to_bf16_kernel(const float* s, unsigned short* d, long long n) {
  long long i = (long long)blockIdx.x * blockDim.x + threadIdx.x;
  if (i < n) d[i] = f2bf(s[i]);
}

// transpose-convert weight [K,N] f32 -> [N,K] bf16 (so GEMM B is k-contiguous)
__global__ void convT_kernel(const float* src, unsigned short* dst, int K, int Nn) {
  long long i = (long long)blockIdx.x * blockDim.x + threadIdx.x;
  if (i < (long long)K * Nn) {
    int k = (int)(i / Nn), n = (int)(i % Nn);
    dst[(long long)n * K + k] = f2bf(src[i]);
  }
}

// LayerNorm over last dim D; one wave per row; optional transposed row gather.
__global__ void ln_kernel(const float* x, const float* gam, const float* bet,
                          unsigned short* out, int rows, int D, int transN) {
  int row  = blockIdx.x * 8 + (threadIdx.x >> 5);
  int lane = threadIdx.x & 31;
  if (row >= rows) return;
  long long src = row;
  if (transN) {
    int i = row / transN, j = row - i * transN;
    src = (long long)j * transN + i;
  }
  const float* xr = x + src * (long long)D;
  float s = 0.0f, s2 = 0.0f;
  for (int j = lane; j < D; j += 32) { float v = xr[j]; s += v; s2 += v * v; }
  for (int o = 16; o > 0; o >>= 1) { s += __shfl_xor(s, o); s2 += __shfl_xor(s2, o); }
  float mean = s / D;
  float var  = s2 / D - mean * mean;
  float inv  = rsqrtf(var + 1e-5f);
  unsigned short* orow = out + (long long)row * D;
  for (int j = lane; j < D; j += 32)
    orow[j] = f2bf((xr[j] - mean) * inv * gam[j] + bet[j]);
}

// gated+masked triangle tensor, written as two d-major planes:
// t0[d][r] = a-half, t1[d][r] = b-half, r = i*N+k; fully contiguous einsum.
__global__ void gate_mask_perm_kernel(const float* p, const float* g, const float* pm,
                                      unsigned short* t0, unsigned short* t1,
                                      long long nn, int P2) {
  long long i = (long long)blockIdx.x * blockDim.x + threadIdx.x;
  if (i >= nn * 2 * P2) return;
  long long c = i / nn, r = i - c * nn;
  long long src = r * 2 * P2 + c;
  float v = p[src] * sigm(g[src]) * pm[r];
  if (c < P2) t0[c * nn + r] = f2bf(v);
  else        t1[(c - P2) * nn + r] = f2bf(v);
}

__global__ void silu_mul_kernel(const float* a, const float* b, unsigned short* o,
                                long long n) {
  long long i = (long long)blockIdx.x * blockDim.x + threadIdx.x;
  if (i < n) { float x = a[i]; o[i] = f2bf(x * sigm(x) * b[i]); }
}

__global__ void sig_gate_kernel(const float* ov, const float* g, unsigned short* o,
                                long long n) {
  long long i = (long long)blockIdx.x * blockDim.x + threadIdx.x;
  if (i < n) o[i] = f2bf(ov[i] * sigm(g[i]));
}

__global__ void combine_add_kernel(float* z, const float* p, const float* g,
                                   long long n) {
  long long i = (long long)blockIdx.x * blockDim.x + threadIdx.x;
  if (i < n) z[i] += p[i] * sigm(g[i]);
}

// Softmax over 256 keys with fused bias.
// mode 0 (triangle): rows = [ci][h][q]; bias = INF*(mask-1) + tb[q,k,h]
// mode 1 (pair-bias): rows = [h][q];    bias = proj[(q*256+k)*H+h] - INF*(1-mask[k])
__global__ void softmax_kernel(const float* sc, unsigned short* pr, int rows,
                               int mode, int i0, const float* mask,
                               const float* bias, int transT, int H) {
  int r = blockIdx.x * blockDim.x + threadIdx.x;
  if (r >= rows) return;
  int i = 0, h, q;
  if (mode == 0) {
    i = i0 + r / (H * 256);
    h = (r >> 8) % H;
    q = r & 255;
  } else {
    h = r >> 8;
    q = r & 255;
  }
  const float* srow = sc + (long long)r * 256;
  unsigned short* prow = pr + (long long)r * 256;
  float mx = -1.0e30f;
  for (int k = 0; k < 256; ++k) {
    float bs;
    if (mode == 0) {
      float mk = transT ? mask[k * 256 + i] : mask[i * 256 + k];
      bs = 1.0e6f * (mk - 1.0f) + bias[(((long long)q << 8) + k) * H + h];
    } else {
      bs = bias[(((long long)q << 8) + k) * H + h] + (1.0f - mask[k]) * -1.0e6f;
    }
    mx = fmaxf(mx, srow[k] + bs);
  }
  float sum = 0.0f;
  for (int k = 0; k < 256; ++k) {
    float bs;
    if (mode == 0) {
      float mk = transT ? mask[k * 256 + i] : mask[i * 256 + k];
      bs = 1.0e6f * (mk - 1.0f) + bias[(((long long)q << 8) + k) * H + h];
    } else {
      bs = bias[(((long long)q << 8) + k) * H + h] + (1.0f - mask[k]) * -1.0e6f;
    }
    float e = __expf(srow[k] + bs - mx);
    sum += e;
    prow[k] = f2bf(e);
  }
  float inv = 1.0f / sum;
  for (int k = 0; k < 256; ++k) prow[k] = f2bf(bf2f(prow[k]) * inv);
}

// ================================ host ======================================
extern "C" void kernel_launch(void* const* d_in, const int* in_sizes, int n_in,
                              void* d_out, int out_size, void* d_ws, size_t ws_size,
                              hipStream_t stream) {
  (void)in_sizes; (void)n_in; (void)out_size; (void)ws_size;
  const int N = 256, P = 128, S = 384, HT = 4, CT = 32, HS = 16, DH = 24;
  const long long NN = (long long)N * N;

  float* s = (float*)d_out;          // [N, S]
  float* z = s + (long long)N * S;   // [N, N, P]
  hipMemcpyAsync(s, d_in[0], sizeof(float) * N * S, hipMemcpyDeviceToDevice, stream);
  hipMemcpyAsync(z, d_in[1], sizeof(float) * NN * P, hipMemcpyDeviceToDevice, stream);
  const float* mask  = (const float*)d_in[2];
  const float* pmask = (const float*)d_in[3];

  char* w8 = (char*)d_ws;
  const size_t MB = 1ull << 20;
  float* F0 = (float*)(w8 + 0 * MB);            // 64 MB f32 scratch
  float* F1 = (float*)(w8 + 64 * MB);           // 64 MB f32 scratch
  float* F2 = (float*)(w8 + 128 * MB);          // 32 MB f32 scratch (gates)
  unsigned short* H0 = (unsigned short*)(w8 + 160 * MB);  // 32 MB bf16
  unsigned short* T0 = H0;                                 // einsum plane a
  unsigned short* T1 = H0 + NN * P;                        // einsum plane b
  unsigned short* XN = (unsigned short*)(w8 + 192 * MB);  // 16 MB bf16 (LN out)
  unsigned short* TN = (unsigned short*)(w8 + 208 * MB);  // 16 MB bf16
  unsigned short* QB = (unsigned short*)(w8 + 224 * MB);  // 16 MB bf16
  unsigned short* KB = (unsigned short*)(w8 + 240 * MB);  // 16 MB bf16
  unsigned short* VB = (unsigned short*)(w8 + 256 * MB);  // 16 MB bf16
  unsigned short* WB0 = (unsigned short*)(w8 + 272 * MB); // weight slots
  unsigned short* WB1 = WB0 + (1 << 20);
  unsigned short* WB2 = WB1 + (1 << 20);
  float* MISC = (float*)(w8 + 280 * MB);        // 8 MB (tri bias / pair-bias proj)

  auto conv = [&](const float* src, unsigned short* dst, long long n) {
    f32_to_bf16_kernel<<<dim3((unsigned)((n + 255) / 256)), 256, 0, stream>>>(src, dst, n);
  };
  auto convT = [&](const float* src, unsigned short* dst, int K, int Nn) {
    long long n = (long long)K * Nn;
    convT_kernel<<<dim3((unsigned)((n + 255) / 256)), 256, 0, stream>>>(src, dst, K, Nn);
  };
  auto gemm = [&](const unsigned short* A, const unsigned short* B, float* C,
                  int M, int Nn, int K, float alpha, int epi, const float* bias,
                  long long aM, long long aK, long long bK, long long bN,
                  long long cM, long long cN, int batch, int bmod,
                  long long aB1, long long aB2, long long bB1, long long bB2,
                  long long cB1, long long cB2) {
    GemmP p;
    p.A = A; p.B = B; p.C = C; p.bias = bias;
    p.aM = aM; p.aK = aK; p.bK = bK; p.bN = bN; p.cM = cM; p.cN = cN;
    p.aB1 = aB1; p.aB2 = aB2; p.bB1 = bB1; p.bB2 = bB2; p.cB1 = cB1; p.cB2 = cB2;
    p.M = M; p.N = Nn; p.K = K; p.bmod = bmod; p.epi = epi; p.alpha = alpha;
    dim3 grid((unsigned)((Nn + 127) / 128), (unsigned)((M + 63) / 64), (unsigned)batch);
    int ap = (aK == 1) ? 0 : 1;   // else aM == 1 by construction
    int bp = (bK == 1) ? 0 : 1;   // else bN == 1 by construction
    if (ap == 0 && bp == 0)      wmma_gemm_kernel<0, 0><<<grid, 256, 0, stream>>>(p);
    else if (ap == 0)            wmma_gemm_kernel<0, 1><<<grid, 256, 0, stream>>>(p);
    else if (bp == 0)            wmma_gemm_kernel<1, 0><<<grid, 256, 0, stream>>>(p);
    else                         wmma_gemm_kernel<1, 1><<<grid, 256, 0, stream>>>(p);
  };

  for (int l = 0; l < 2; ++l) {
    auto PPf = [&](int idx, long long per) {
      return (const float*)d_in[idx] + (long long)l * per;
    };

    // ------------------- triangle multiplication (outgoing, incoming) -------
    for (int dir = 0; dir < 2; ++dir) {
      int pb = 4 + dir * 8;          // tmo_:4..11  tmi_:12..19
      int outgoing = (dir == 0);
      ln_kernel<<<dim3((unsigned)((NN + 7) / 8)), 256, 0, stream>>>(
          z, PPf(pb + 0, P), PPf(pb + 1, P), XN, (int)NN, P, 0);
      convT(PPf(pb + 2, P * 2 * P), WB0, P, 2 * P);
      gemm(XN, WB0, F0, (int)NN, 2 * P, P, 1.f, 0, nullptr,
           P, 1, 1, P, 2 * P, 1, 1, 1, 0, 0, 0, 0, 0, 0);
      convT(PPf(pb + 3, P * 2 * P), WB0, P, 2 * P);
      gemm(XN, WB0, F1, (int)NN, 2 * P, P, 1.f, 0, nullptr,
           P, 1, 1, P, 2 * P, 1, 1, 1, 0, 0, 0, 0, 0, 0);
      gate_mask_perm_kernel<<<dim3((unsigned)((NN * 2 * P + 255) / 256)), 256, 0,
                              stream>>>(F0, F1, pmask, T0, T1, NN, P);
      // einsum: batched over d (grid.z = 128), t -> F1 [NN, P]
      if (outgoing)  // out[i,j] = sum_k T0[d][i*N+k] * T1[d][j*N+k]
        gemm(T0, T1, F1, N, N, N, 1.f, 0, nullptr,
             N, 1,                // A: i, k (k-contig)
             1, N,                // B: k, j (k-contig)
             (long long)N * P, P, // C: i, j
             P, 1, NN, 0, NN, 0, 1, 0);
      else           // out[i,j] = sum_k T0[d][k*N+i] * T1[d][k*N+j]
        gemm(T0, T1, F1, N, N, N, 1.f, 0, nullptr,
             1, N,                // A: m-contig
             N, 1,                // B: n-contig
             (long long)N * P, P,
             P, 1, NN, 0, NN, 0, 1, 0);
      ln_kernel<<<dim3((unsigned)((NN + 7) / 8)), 256, 0, stream>>>(
          F1, PPf(pb + 4, P), PPf(pb + 5, P), TN, (int)NN, P, 0);
      convT(PPf(pb + 6, P * P), WB0, P, P);
      gemm(TN, WB0, F0, (int)NN, P, P, 1.f, 0, nullptr,
           P, 1, 1, P, P, 1, 1, 1, 0, 0, 0, 0, 0, 0);
      convT(PPf(pb + 7, P * P), WB0, P, P);
      gemm(XN, WB0, F2, (int)NN, P, P, 1.f, 0, nullptr,
           P, 1, 1, P, P, 1, 1, 1, 0, 0, 0, 0, 0, 0);
      combine_add_kernel<<<dim3((unsigned)((NN * P + 255) / 256)), 256, 0, stream>>>(
          z, F0, F2, NN * P);
    }

    // ------------------- triangle attention (starting, ending) --------------
    for (int dir = 0; dir < 2; ++dir) {
      int pb = 20 + dir * 8;         // tas_:20..27  tae_:28..35
      int T = dir;                   // ending uses transposed view of z
      ln_kernel<<<dim3((unsigned)((NN + 7) / 8)), 256, 0, stream>>>(
          z, PPf(pb + 0, P), PPf(pb + 1, P), XN, (int)NN, P, T ? N : 0);
      convT(PPf(pb + 2, P * HT), WB0, P, HT);
      gemm(XN, WB0, MISC, (int)NN, HT, P, 1.f, 0, nullptr,
           P, 1, 1, P, HT, 1, 1, 1, 0, 0, 0, 0, 0, 0);           // tri bias
      convT(PPf(pb + 3, P * P), WB0, P, P);
      gemm(XN, WB0, F0, (int)NN, P, P, 1.f, 0, nullptr,
           P, 1, 1, P, P, 1, 1, 1, 0, 0, 0, 0, 0, 0);
      conv(F0, QB, NN * P);
      convT(PPf(pb + 4, P * P), WB0, P, P);
      gemm(XN, WB0, F0, (int)NN, P, P, 1.f, 0, nullptr,
           P, 1, 1, P, P, 1, 1, 1, 0, 0, 0, 0, 0, 0);
      conv(F0, KB, NN * P);
      convT(PPf(pb + 5, P * P), WB0, P, P);
      gemm(XN, WB0, F0, (int)NN, P, P, 1.f, 0, nullptr,
           P, 1, 1, P, P, 1, 1, 1, 0, 0, 0, 0, 0, 0);
      conv(F0, VB, NN * P);
      convT(PPf(pb + 6, P * P), WB0, P, P);
      gemm(XN, WB0, F2, (int)NN, P, P, 1.f, 0, nullptr,
           P, 1, 1, P, P, 1, 1, 1, 0, 0, 0, 0, 0, 0);            // gate f32
      const int CH = 64;  // i-chunk
      for (int i0 = 0; i0 < N; i0 += CH) {
        // scores[ci][h][q][k] -> F0
        gemm(QB + (long long)i0 * N * P, KB + (long long)i0 * N * P, F0,
             N, N, CT, 0.1767766953f, 0, nullptr,
             P, 1,                    // A: q-row stride, c stride (k-contig)
             1, P,                    // B: c stride (k-contig), key stride
             N, 1,                    // C: q, k
             CH * HT, HT,
             (long long)N * P, CT,    // A batch: ci, h
             (long long)N * P, CT,    // B batch: ci, h
             (long long)HT * N * N, (long long)N * N);
        softmax_kernel<<<dim3((unsigned)((CH * HT * N + 255) / 256)), 256, 0, stream>>>(
            F0, H0, CH * HT * N, 0, i0, pmask, MISC, T, HT);
        // o[ci][q][h][c] -> F1 (full [NN, HT*CT] layout)
        gemm(H0, VB + (long long)i0 * N * P, F1 + (long long)i0 * N * P,
             N, CT, N, 1.f, 0, nullptr,
             N, 1,                    // A: q, k (k-contig)
             P, 1,                    // B: key stride, c stride (n-contig)
             P, 1,                    // C: q stride, c stride
             CH * HT, HT,
             (long long)HT * N * N, (long long)N * N,
             (long long)N * P, CT,
             (long long)N * P, CT);
      }
      sig_gate_kernel<<<dim3((unsigned)((NN * P + 255) / 256)), 256, 0, stream>>>(
          F1, F2, H0, NN * P);
      convT(PPf(pb + 7, P * P), WB0, P, P);
      if (!T)
        gemm(H0, WB0, z, (int)NN, P, P, 1.f, 1, nullptr,
             P, 1, 1, P, P, 1, 1, 1, 0, 0, 0, 0, 0, 0);
      else  // transposed residual write: z[j,i] += o[i,j] @ W
        gemm(H0, WB0, z, N, P, P, 1.f, 1, nullptr,
             P, 1, 1, P, (long long)N * P, 1,
             N, 1, (long long)N * P, 0, 0, 0, P, 0);
    }

    // ------------------- transitions -----------------------------------------
    auto transition = [&](float* x, int rows, int d, int pb) {
      ln_kernel<<<dim3((unsigned)((rows + 7) / 8)), 256, 0, stream>>>(
          x, PPf(pb + 0, d), PPf(pb + 1, d), XN, rows, d, 0);
      convT(PPf(pb + 2, (long long)d * 4 * d), WB0, d, 4 * d);
      convT(PPf(pb + 3, (long long)d * 4 * d), WB1, d, 4 * d);
      convT(PPf(pb + 4, (long long)d * 4 * d), WB2, 4 * d, d);
      const int R = 16384;
      for (int r0 = 0; r0 < rows; r0 += R) {
        int cm = (rows - r0 < R) ? (rows - r0) : R;
        gemm(XN + (long long)r0 * d, WB0, F0, cm, 4 * d, d, 1.f, 0, nullptr,
             d, 1, 1, d, 4 * d, 1, 1, 1, 0, 0, 0, 0, 0, 0);
        gemm(XN + (long long)r0 * d, WB1, F1, cm, 4 * d, d, 1.f, 0, nullptr,
             d, 1, 1, d, 4 * d, 1, 1, 1, 0, 0, 0, 0, 0, 0);
        silu_mul_kernel<<<dim3((unsigned)(((long long)cm * 4 * d + 255) / 256)), 256,
                          0, stream>>>(F0, F1, H0, (long long)cm * 4 * d);
        gemm(H0, WB2, x + (long long)r0 * d, cm, d, 4 * d, 1.f, 1, nullptr,
             4 * d, 1, 1, 4 * d, d, 1, 1, 1, 0, 0, 0, 0, 0, 0);
      }
    };
    transition(z, (int)NN, P, 36);

    // ------------------- attention with pair bias on s (46..57) --------------
    ln_kernel<<<dim3((unsigned)((NN + 7) / 8)), 256, 0, stream>>>(
        z, PPf(53, P), PPf(54, P), XN, (int)NN, P, 0);
    convT(PPf(55, P * HS), WB0, P, HS);
    gemm(XN, WB0, MISC, (int)NN, HS, P, 1.f, 0, nullptr,
         P, 1, 1, P, HS, 1, 1, 1, 0, 0, 0, 0, 0, 0);            // pair bias proj
    ln_kernel<<<dim3((unsigned)((N + 7) / 8)), 256, 0, stream>>>(
        s, PPf(46, S), PPf(47, S), TN, N, S, 0);
    convT(PPf(48, (long long)S * S), WB0, S, S);
    gemm(TN, WB0, F0, N, S, S, 1.f, 0, PPf(49, S),
         S, 1, 1, S, S, 1, 1, 1, 0, 0, 0, 0, 0, 0);
    conv(F0, QB, (long long)N * S);
    convT(PPf(50, (long long)S * S), WB0, S, S);
    gemm(TN, WB0, F0, N, S, S, 1.f, 0, nullptr,
         S, 1, 1, S, S, 1, 1, 1, 0, 0, 0, 0, 0, 0);
    conv(F0, KB, (long long)N * S);
    convT(PPf(51, (long long)S * S), WB0, S, S);
    gemm(TN, WB0, F0, N, S, S, 1.f, 0, nullptr,
         S, 1, 1, S, S, 1, 1, 1, 0, 0, 0, 0, 0, 0);
    conv(F0, VB, (long long)N * S);
    convT(PPf(52, (long long)S * S), WB0, S, S);
    gemm(TN, WB0, F2, N, S, S, 1.f, 0, nullptr,
         S, 1, 1, S, S, 1, 1, 1, 0, 0, 0, 0, 0, 0);             // gate f32
    // scores[h][i][j]  (K = 24 handled by K-tail epilogue)
    gemm(QB, KB, F0, N, N, DH, 0.2041241452f, 0, nullptr,
         S, 1, 1, S, N, 1,
         HS, HS, 0, DH, 0, DH, 0, (long long)N * N);
    softmax_kernel<<<dim3((unsigned)((HS * N + 255) / 256)), 256, 0, stream>>>(
        F0, H0, HS * N, 1, 0, mask, MISC, 0, HS);
    gemm(H0, VB, F1, N, DH, N, 1.f, 0, nullptr,
         N, 1, S, 1, S, 1,
         HS, HS, 0, (long long)N * N, 0, DH, 0, DH);
    sig_gate_kernel<<<dim3((unsigned)(((long long)N * S + 255) / 256)), 256, 0, stream>>>(
        F1, F2, H0, (long long)N * S);
    convT(PPf(56, (long long)S * S), WB0, S, S);
    gemm(H0, WB0, s, N, S, S, 1.f, 1, PPf(57, S),
         S, 1, 1, S, S, 1, 1, 1, 0, 0, 0, 0, 0, 0);

    // ------------------- s transition (41..45) -------------------------------
    transition(s, N, S, 41);
  }
}
